// DistanceAwareTransformerEncoderLayer_51049981280442
// MI455X (gfx1250) — compile-verified, matrix operand order checked by reading, not run on hardware
//
#include <hip/hip_runtime.h>
#include <hip/hip_bf16.h>

// ---------------- problem constants ----------------
#define BB   4
#define SS   1024
#define DD   1024
#define HH   16
#define HD   64
#define DFF  4096
#define NTOK (BB * SS)          // 4096

typedef __bf16 bf16_t;
typedef __attribute__((ext_vector_type(16))) __bf16 v16bf;
typedef __attribute__((ext_vector_type(8)))  __bf16 v8bf;
typedef __attribute__((ext_vector_type(8)))  float  v8f;

union Frag { v16bf v; v8bf h[2]; };

__device__ __forceinline__ bf16_t f2bf(float f) {
    union { float f; unsigned u; } a; a.f = f;
    unsigned r = a.u + 0x7fffu + ((a.u >> 16) & 1u);
    union { unsigned short s; bf16_t b; } c; c.s = (unsigned short)(r >> 16);
    return c.b;
}

// A/B fragment for v_wmma_f32_16x16x32_bf16 (16-bit A layout, ISA 7.12.2):
// lane half h: elems 0..7 -> K = kb + 8h + 0..7 ; elems 8..15 -> K = kb+16+8h+0..7
__device__ __forceinline__ Frag load_frag(const bf16_t* rowbase, int kb, int lhalf) {
    Frag f;
    f.h[0] = *(const v8bf*)(rowbase + kb + lhalf * 8);
    f.h[1] = *(const v8bf*)(rowbase + kb + 16 + lhalf * 8);
    return f;
}

__device__ __forceinline__ v8f wmma_bf16(const Frag& a, const Frag& b, v8f c) {
    return __builtin_amdgcn_wmma_f32_16x16x32_bf16(
        false, a.v, false, b.v, (short)0, c, false, false);
}

// ---------------- conversion kernels ----------------
__global__ __launch_bounds__(256) void k_f32_to_bf16(const float* __restrict__ in,
                                                     bf16_t* __restrict__ out, int n) {
    int i = blockIdx.x * 256 + threadIdx.x;
    if (i < n) out[i] = f2bf(in[i]);
}

// W [K][N] f32 -> Wt [N][K] bf16
__global__ __launch_bounds__(256) void k_transpose_w(const float* __restrict__ W,
                                                     bf16_t* __restrict__ Wt, int K, int N) {
    int i = blockIdx.x * 256 + threadIdx.x;
    if (i >= K * N) return;
    int k = i / N, n = i - k * N;
    Wt[(size_t)n * K + k] = f2bf(W[i]);
}

// ---------------- generic 32x32-per-wave WMMA GEMM core ----------------
struct TileCtx { int m0, n0, col, lhalf; v8f acc[2][2]; };

__device__ __forceinline__ bool gemm32x32(const bf16_t* __restrict__ A,
                                          const bf16_t* __restrict__ Bt,
                                          int M, int N, int K, TileCtx& t) {
    int wave = (blockIdx.x * 256 + threadIdx.x) >> 5;
    int tn = N >> 5;
    if (wave >= (M >> 5) * tn) return false;            // whole-wave exit: EXEC stays full
    t.m0 = (wave / tn) << 5;
    t.n0 = (wave % tn) << 5;
    int lane = threadIdx.x & 31;
    t.col = lane & 15;
    t.lhalf = lane >> 4;
    const bf16_t* a0 = A + (size_t)(t.m0 + t.col) * K;
    const bf16_t* a1 = a0 + (size_t)16 * K;
    const bf16_t* b0 = Bt + (size_t)(t.n0 + t.col) * K;
    const bf16_t* b1 = b0 + (size_t)16 * K;
    v8f z = {0.f, 0.f, 0.f, 0.f, 0.f, 0.f, 0.f, 0.f};
    t.acc[0][0] = z; t.acc[0][1] = z; t.acc[1][0] = z; t.acc[1][1] = z;
    for (int kb = 0; kb < K; kb += 32) {
        Frag fa0 = load_frag(a0, kb, t.lhalf);
        Frag fa1 = load_frag(a1, kb, t.lhalf);
        Frag fb0 = load_frag(b0, kb, t.lhalf);
        Frag fb1 = load_frag(b1, kb, t.lhalf);
        if (kb + 32 < K) {                               // global_prefetch_b8 path
            __builtin_prefetch(a0 + kb + 64, 0, 1);
            __builtin_prefetch(b0 + kb + 64, 0, 1);
        }
        t.acc[0][0] = wmma_bf16(fa0, fb0, t.acc[0][0]);
        t.acc[0][1] = wmma_bf16(fa0, fb1, t.acc[0][1]);
        t.acc[1][0] = wmma_bf16(fa1, fb0, t.acc[1][0]);
        t.acc[1][1] = wmma_bf16(fa1, fb1, t.acc[1][1]);
    }
    return true;
}

// ---------------- Q/K/V projection ----------------
// mode 0: q -> [B,H,S,HD] scaled; mode 1: k -> [B,H,S,HD]; mode 2: v -> vT [B,H,HD,S]
__global__ __launch_bounds__(256) void k_proj_qkv(const bf16_t* __restrict__ X,
                                                  const bf16_t* __restrict__ Wt,
                                                  const float* __restrict__ bias,
                                                  bf16_t* __restrict__ out,
                                                  int mode, float scale) {
    TileCtx t;
    if (!gemm32x32(X, Wt, NTOK, DD, DD, t)) return;
#pragma unroll
    for (int mt = 0; mt < 2; ++mt)
#pragma unroll
        for (int nt = 0; nt < 2; ++nt) {
            int n = t.n0 + nt * 16 + t.col;
            int h = n >> 6, hd = n & 63;
            float bv = bias[n];
#pragma unroll
            for (int r = 0; r < 8; ++r) {
                int row = t.m0 + mt * 16 + r + 8 * t.lhalf;
                int b = row >> 10, s = row & 1023;
                float val = (t.acc[mt][nt][r] + bv) * scale;
                size_t idx;
                if (mode == 2) idx = (((size_t)(b * HH + h) * HD + hd) << 10) + s;
                else           idx = (((size_t)(b * HH + h) << 10) + s) * HD + hd;
                out[idx] = f2bf(val);
            }
        }
}

// ---------------- GEMM + bias + residual (f32 out) ----------------
__global__ __launch_bounds__(256) void k_gemm_bias_res_f32(const bf16_t* __restrict__ A,
                                                           const bf16_t* __restrict__ Bt,
                                                           const float* __restrict__ bias,
                                                           const float* __restrict__ res,
                                                           float* __restrict__ out,
                                                           int M, int N, int K) {
    TileCtx t;
    if (!gemm32x32(A, Bt, M, N, K, t)) return;
#pragma unroll
    for (int mt = 0; mt < 2; ++mt)
#pragma unroll
        for (int nt = 0; nt < 2; ++nt) {
            int n = t.n0 + nt * 16 + t.col;
            float bv = bias[n];
#pragma unroll
            for (int r = 0; r < 8; ++r) {
                int row = t.m0 + mt * 16 + r + 8 * t.lhalf;
                size_t idx = (size_t)row * N + n;
                out[idx] = t.acc[mt][nt][r] + bv + res[idx];
            }
        }
}

// ---------------- GEMM + bias + ReLU (bf16 out) ----------------
__global__ __launch_bounds__(256) void k_gemm_bias_relu_bf16(const bf16_t* __restrict__ A,
                                                             const bf16_t* __restrict__ Bt,
                                                             const float* __restrict__ bias,
                                                             bf16_t* __restrict__ out,
                                                             int M, int N, int K) {
    TileCtx t;
    if (!gemm32x32(A, Bt, M, N, K, t)) return;
#pragma unroll
    for (int mt = 0; mt < 2; ++mt)
#pragma unroll
        for (int nt = 0; nt < 2; ++nt) {
            int n = t.n0 + nt * 16 + t.col;
            float bv = bias[n];
#pragma unroll
            for (int r = 0; r < 8; ++r) {
                int row = t.m0 + mt * 16 + r + 8 * t.lhalf;
                out[(size_t)row * N + n] = f2bf(fmaxf(t.acc[mt][nt][r] + bv, 0.f));
            }
        }
}

// ---------------- flash attention (1 wave = 16 queries, full S sweep) ----------------
__global__ __launch_bounds__(256) void k_attention(const bf16_t* __restrict__ q,
                                                   const bf16_t* __restrict__ k,
                                                   const bf16_t* __restrict__ vT,
                                                   const float* __restrict__ distances,
                                                   const float* __restrict__ dist_scale_p,
                                                   bf16_t* __restrict__ ao) {
    __shared__ bf16_t Plds[8][16][32];                  // 1KB per wave
    int wave  = (blockIdx.x * 256 + threadIdx.x) >> 5;  // 0..4095
    int wIn   = threadIdx.x >> 5;
    int lane  = threadIdx.x & 31;
    int col   = lane & 15, lhalf = lane >> 4;
    int qb = wave & 63;          // S/16
    int bh = wave >> 6;          // 0..63
    int b  = bh >> 4;
    int h  = bh & 15;
    int q0 = qb << 4;
    float dsc = fabsf(dist_scale_p[0]);

    const bf16_t* qrow = q + ((size_t)bh * SS + q0 + col) * HD;
    Frag aq0 = load_frag(qrow, 0, lhalf);
    Frag aq1 = load_frag(qrow, 32, lhalf);

    v8f z = {0.f, 0.f, 0.f, 0.f, 0.f, 0.f, 0.f, 0.f};
    v8f acco[4] = {z, z, z, z};
    float mrow[8], lrow[8];
#pragma unroll
    for (int r = 0; r < 8; ++r) { mrow[r] = -1e30f; lrow[r] = 0.f; }

    const float* drow = distances + ((size_t)b * SS + q0) * SS;

    for (int kb = 0; kb < SS; kb += 32) {
        // ---- scores: Q(16x64) x K^T(64x32) ----
        const bf16_t* krow0 = k + ((size_t)bh * SS + kb + col) * HD;
        const bf16_t* krow1 = krow0 + (size_t)16 * HD;
        Frag b00 = load_frag(krow0, 0, lhalf), b01 = load_frag(krow0, 32, lhalf);
        Frag b10 = load_frag(krow1, 0, lhalf), b11 = load_frag(krow1, 32, lhalf);
        v8f s0 = z, s1 = z;
        s0 = wmma_bf16(aq0, b00, s0); s0 = wmma_bf16(aq1, b01, s0);
        s1 = wmma_bf16(aq0, b10, s1); s1 = wmma_bf16(aq1, b11, s1);

        // ---- bias + online softmax (row stats live across a 16-lane group) ----
#pragma unroll
        for (int r = 0; r < 8; ++r) {
            int row = r + 8 * lhalf;
            const float* dp = drow + (size_t)row * SS + kb + col;
            float t0 = s0[r] + (-__logf(dp[0]  + 1e-9f)) * dsc;
            float t1 = s1[r] + (-__logf(dp[16] + 1e-9f)) * dsc;
            float mx = fmaxf(t0, t1);
            mx = fmaxf(mx, __shfl_xor(mx, 1, 32));
            mx = fmaxf(mx, __shfl_xor(mx, 2, 32));
            mx = fmaxf(mx, __shfl_xor(mx, 4, 32));
            mx = fmaxf(mx, __shfl_xor(mx, 8, 32));
            float mnew  = fmaxf(mrow[r], mx);
            float alpha = __expf(mrow[r] - mnew);
            float p0 = __expf(t0 - mnew);
            float p1 = __expf(t1 - mnew);
            float ps = p0 + p1;
            ps += __shfl_xor(ps, 1, 32);
            ps += __shfl_xor(ps, 2, 32);
            ps += __shfl_xor(ps, 4, 32);
            ps += __shfl_xor(ps, 8, 32);
            lrow[r] = lrow[r] * alpha + ps;
            mrow[r] = mnew;
#pragma unroll
            for (int nt = 0; nt < 4; ++nt) acco[nt][r] *= alpha;
            Plds[wIn][row][col]      = f2bf(p0);
            Plds[wIn][row][col + 16] = f2bf(p1);
        }
        // C-layout -> A-layout through LDS (same-wave DS ops are in-order)
        asm volatile("s_wait_dscnt 0" ::: "memory");
        Frag ap;
        ap.h[0] = *(const v8bf*)&Plds[wIn][col][8 * lhalf];
        ap.h[1] = *(const v8bf*)&Plds[wIn][col][16 + 8 * lhalf];
        asm volatile("" ::: "memory");                  // keep next iter's stores below the reads

        // ---- P(16x32) x V(32x64), V pre-transposed so B rows are contiguous ----
#pragma unroll
        for (int nt = 0; nt < 4; ++nt) {
            const bf16_t* vrow = vT + ((size_t)bh * HD + nt * 16 + col) * SS;
            Frag bv = load_frag(vrow, kb, lhalf);
            acco[nt] = wmma_bf16(ap, bv, acco[nt]);
        }
    }

    // ---- normalize + write ao [B,S,D] bf16 ----
#pragma unroll
    for (int r = 0; r < 8; ++r) {
        int row = q0 + r + 8 * lhalf;
        float inv = 1.0f / lrow[r];
        size_t base = ((size_t)b * SS + row) * DD + h * HD;
#pragma unroll
        for (int nt = 0; nt < 4; ++nt)
            ao[base + nt * 16 + col] = f2bf(acco[nt][r] * inv);
    }
}

// ---------------- LayerNorm: 1 block per row of 1024 ----------------
__global__ __launch_bounds__(256) void k_layernorm(const float* __restrict__ y,
                                                   const float* __restrict__ g,
                                                   const float* __restrict__ bt,
                                                   float* __restrict__ outf,
                                                   bf16_t* __restrict__ outbf) {
    __shared__ float red[256];
    int tid = threadIdx.x;
    const float* yr = y + (size_t)blockIdx.x * DD;
    float v[4]; float s = 0.f;
#pragma unroll
    for (int i = 0; i < 4; ++i) { v[i] = yr[tid + 256 * i]; s += v[i]; }
    red[tid] = s; __syncthreads();
    for (int off = 128; off > 0; off >>= 1) { if (tid < off) red[tid] += red[tid + off]; __syncthreads(); }
    float mu = red[0] * (1.f / DD);
    __syncthreads();
    float s2 = 0.f;
#pragma unroll
    for (int i = 0; i < 4; ++i) { float d = v[i] - mu; s2 += d * d; }
    red[tid] = s2; __syncthreads();
    for (int off = 128; off > 0; off >>= 1) { if (tid < off) red[tid] += red[tid + off]; __syncthreads(); }
    float rs = rsqrtf(red[0] * (1.f / DD) + 1e-5f);
#pragma unroll
    for (int i = 0; i < 4; ++i) {
        int idx = tid + 256 * i;
        float o = (v[i] - mu) * rs * g[idx] + bt[idx];
        if (outf)  outf[(size_t)blockIdx.x * DD + idx] = o;
        if (outbf) outbf[(size_t)blockIdx.x * DD + idx] = f2bf(o);
    }
}

// ---------------- host-side orchestration ----------------
extern "C" void kernel_launch(void* const* d_in, const int* in_sizes, int n_in,
                              void* d_out, int out_size, void* d_ws, size_t ws_size,
                              hipStream_t stream) {
    const float* src  = (const float*)d_in[0];
    const float* dist = (const float*)d_in[1];
    const float* Wq = (const float*)d_in[2];  const float* bq = (const float*)d_in[3];
    const float* Wk = (const float*)d_in[4];  const float* bk = (const float*)d_in[5];
    const float* Wv = (const float*)d_in[6];  const float* bv = (const float*)d_in[7];
    const float* Wo = (const float*)d_in[8];  const float* bo = (const float*)d_in[9];
    const float* dsc = (const float*)d_in[10];
    const float* W1 = (const float*)d_in[11]; const float* b1 = (const float*)d_in[12];
    const float* W2 = (const float*)d_in[13]; const float* b2 = (const float*)d_in[14];
    const float* g1 = (const float*)d_in[15]; const float* be1 = (const float*)d_in[16];
    const float* g2 = (const float*)d_in[17]; const float* be2 = (const float*)d_in[18];
    (void)in_sizes; (void)n_in; (void)out_size; (void)ws_size;

    char* w = (char*)d_ws;
    const size_t MB = 1u << 20;
    bf16_t* WtQ = (bf16_t*)w; w += 2 * MB;
    bf16_t* WtK = (bf16_t*)w; w += 2 * MB;
    bf16_t* WtV = (bf16_t*)w; w += 2 * MB;
    bf16_t* WtO = (bf16_t*)w; w += 2 * MB;
    bf16_t* Wt1 = (bf16_t*)w; w += 8 * MB;   // [DFF][D]
    bf16_t* Wt2 = (bf16_t*)w; w += 8 * MB;   // [D][DFF]
    bf16_t* Xbf = (bf16_t*)w; w += 8 * MB;
    bf16_t* qb_ = (bf16_t*)w; w += 8 * MB;   // [B,H,S,HD]
    bf16_t* kb_ = (bf16_t*)w; w += 8 * MB;   // [B,H,S,HD]
    bf16_t* vT_ = (bf16_t*)w; w += 8 * MB;   // [B,H,HD,S]
    bf16_t* aob = (bf16_t*)w; w += 8 * MB;   // [B,S,D]
    float*  y1  = (float*)w;  w += 16 * MB;
    bf16_t* xbf = (bf16_t*)w; w += 8 * MB;
    float*  xf  = (float*)w;  w += 16 * MB;
    bf16_t* hbf = (bf16_t*)w; w += 32 * MB;  // [NTOK][DFF]
    float*  y2  = (float*)w;  w += 16 * MB;

    dim3 blk(256);
    // 1) conversions
    k_f32_to_bf16<<<(NTOK * DD) / 256, blk, 0, stream>>>(src, Xbf, NTOK * DD);
    k_transpose_w<<<(DD * DD) / 256,  blk, 0, stream>>>(Wq, WtQ, DD, DD);
    k_transpose_w<<<(DD * DD) / 256,  blk, 0, stream>>>(Wk, WtK, DD, DD);
    k_transpose_w<<<(DD * DD) / 256,  blk, 0, stream>>>(Wv, WtV, DD, DD);
    k_transpose_w<<<(DD * DD) / 256,  blk, 0, stream>>>(Wo, WtO, DD, DD);
    k_transpose_w<<<(DD * DFF) / 256, blk, 0, stream>>>(W1, Wt1, DD, DFF);
    k_transpose_w<<<(DFF * DD) / 256, blk, 0, stream>>>(W2, Wt2, DFF, DD);

    // 2) Q/K/V projections: (4096/32)*(1024/32)=4096 waves -> 512 blocks
    const int projBlocks = ((NTOK / 32) * (DD / 32) * 32) / 256;
    const float qscale = 0.125f;  // HD^-0.5
    k_proj_qkv<<<projBlocks, blk, 0, stream>>>(Xbf, WtQ, bq, qb_, 0, qscale);
    k_proj_qkv<<<projBlocks, blk, 0, stream>>>(Xbf, WtK, bk, kb_, 1, 1.0f);
    k_proj_qkv<<<projBlocks, blk, 0, stream>>>(Xbf, WtV, bv, vT_, 2, 1.0f);

    // 3) attention: B*H*(S/16)=4096 waves -> 512 blocks
    k_attention<<<(BB * HH * (SS / 16) * 32) / 256, blk, 0, stream>>>(qb_, kb_, vT_, dist, dsc, aob);

    // 4) O projection + residual, LN1
    k_gemm_bias_res_f32<<<projBlocks, blk, 0, stream>>>(aob, WtO, bo, src, y1, NTOK, DD, DD);
    k_layernorm<<<NTOK, blk, 0, stream>>>(y1, g1, be1, xf, xbf);

    // 5) FFN
    const int ffn1Blocks = ((NTOK / 32) * (DFF / 32) * 32) / 256;  // 2048
    k_gemm_bias_relu_bf16<<<ffn1Blocks, blk, 0, stream>>>(xbf, Wt1, b1, hbf, NTOK, DFF, DD);
    k_gemm_bias_res_f32<<<projBlocks, blk, 0, stream>>>(hbf, Wt2, b2, xf, y2, NTOK, DD, DFF);

    // 6) final LN -> d_out (f32)
    k_layernorm<<<NTOK, blk, 0, stream>>>(y2, g2, be2, (float*)d_out, (bf16_t*)nullptr);
}
